// DenseTf_83726092468397
// MI455X (gfx1250) — matrix operand order repaired and checked by reading, not compiled, not run on hardware
//
#include <hip/hip_runtime.h>

#define N_ROWS 8192   // N  (batch rows)
#define K_DIM  2048   // D_IN
#define N_COLS 2048   // D_OUT

typedef __attribute__((ext_vector_type(16))) _Float16 v16h;
typedef __attribute__((ext_vector_type(8)))  _Float16 v8h;
typedef __attribute__((ext_vector_type(8)))  float    v8f;

union V16 { v16h v; v8h h[2]; };

// ---------------------------------------------------------------------------
// Kernel 1: Wt[n][k] = sign(W[k][n]) as f16, LDS-tiled transpose (32x32 tiles)
// ---------------------------------------------------------------------------
__global__ __launch_bounds__(256)
void binarize_transpose_w(const float* __restrict__ w, _Float16* __restrict__ wt) {
    __shared__ float tile[32][33];
    const int bn = blockIdx.x * 32;   // n tile base
    const int bk = blockIdx.y * 32;   // k tile base

    // read W row-major [K][N]: coalesced along n
    #pragma unroll
    for (int i = 0; i < 4; ++i) {
        int kk = threadIdx.y + i * 8;
        tile[kk][threadIdx.x] = w[(size_t)(bk + kk) * N_COLS + (bn + threadIdx.x)];
    }
    __syncthreads();

    // write Wt row-major [N][K]: coalesced along k
    #pragma unroll
    for (int i = 0; i < 4; ++i) {
        int nn = threadIdx.y + i * 8;
        float v = tile[threadIdx.x][nn];
        _Float16 s = (v > 0.0f) ? (_Float16)1.0f
                   : ((v < 0.0f) ? (_Float16)(-1.0f) : (_Float16)0.0f);
        wt[(size_t)(bn + nn) * K_DIM + (bk + threadIdx.x)] = s;
    }
}

// ---------------------------------------------------------------------------
// Kernel 2: split fp32 A into hi/lo f16 pair (hi + lo ~ 22-bit mantissa of x)
// ---------------------------------------------------------------------------
__global__ __launch_bounds__(256)
void split_a(const float* __restrict__ a, _Float16* __restrict__ hi,
             _Float16* __restrict__ lo, int total) {
    int idx = blockIdx.x * blockDim.x + threadIdx.x;
    if (idx < total) {
        float x = a[idx];
        _Float16 h = (_Float16)x;
        hi[idx] = h;
        lo[idx] = (_Float16)(x - (float)h);
    }
}

// ---------------------------------------------------------------------------
// Kernel 3: WMMA GEMM. Block = 256 threads = 8 waves; block tile 64(M)x256(N).
// Wave tile 32x64 = 2x4 subtiles of 16x16. K-step = 32.
// acc += Ahi_tile @ B_tile ; acc += Alo_tile @ B_tile   (weights are exact +/-1)
// ---------------------------------------------------------------------------
__global__ __launch_bounds__(256)
void gemm_wmma(const _Float16* __restrict__ Ahi, const _Float16* __restrict__ Alo,
               const _Float16* __restrict__ Wt,  const float* __restrict__ bias,
               float* __restrict__ out) {
    const int lane  = threadIdx.x & 31;
    const int wv    = threadIdx.x >> 5;   // 0..7
    const int waveM = wv >> 2;            // 0..1
    const int waveN = wv & 3;             // 0..3
    const int Mbase = blockIdx.y * 64  + waveM * 32;
    const int Nbase = blockIdx.x * 256 + waveN * 64;

    const int lhalf = lane >> 4;          // 0 or 1 (half-wave)
    const int l15   = lane & 15;

    v8f acc[2][4];
    #pragma unroll
    for (int mt = 0; mt < 2; ++mt)
        #pragma unroll
        for (int nt = 0; nt < 4; ++nt)
            acc[mt][nt] = (v8f){0.f, 0.f, 0.f, 0.f, 0.f, 0.f, 0.f, 0.f};

    // Per-lane base pointers (K layout per CDNA5 ISA 16-bit A / B fragments):
    //   A lane: row m = Mbase+mt*16+l15 ; halves [k0+lhalf*8 .. +7] and [+16 .. +23]
    //   B lane: row n = Nbase+nt*16+l15 ; halves [k0+lhalf*8 .. +7] and [+16 .. +23]
    const size_t aRow0 = (size_t)(Mbase + 0  + l15) * K_DIM + lhalf * 8;
    const size_t aRow1 = (size_t)(Mbase + 16 + l15) * K_DIM + lhalf * 8;

    for (int k0 = 0; k0 < K_DIM; k0 += 32) {
        V16 ah[2], al[2], bf[4];

        const _Float16* pa0 = Ahi + aRow0 + k0;
        const _Float16* pa1 = Ahi + aRow1 + k0;
        const _Float16* pl0 = Alo + aRow0 + k0;
        const _Float16* pl1 = Alo + aRow1 + k0;
        ah[0].h[0] = *(const v8h*)(pa0);      ah[0].h[1] = *(const v8h*)(pa0 + 16);
        ah[1].h[0] = *(const v8h*)(pa1);      ah[1].h[1] = *(const v8h*)(pa1 + 16);
        al[0].h[0] = *(const v8h*)(pl0);      al[0].h[1] = *(const v8h*)(pl0 + 16);
        al[1].h[0] = *(const v8h*)(pl1);      al[1].h[1] = *(const v8h*)(pl1 + 16);

        #pragma unroll
        for (int nt = 0; nt < 4; ++nt) {
            const _Float16* pb =
                Wt + (size_t)(Nbase + nt * 16 + l15) * K_DIM + k0 + lhalf * 8;
            bf[nt].h[0] = *(const v8h*)(pb);
            bf[nt].h[1] = *(const v8h*)(pb + 16);
        }

        // Software prefetch of next K-tile -> global_prefetch_b8 (gfx1250)
        if (k0 + 32 < K_DIM) {
            __builtin_prefetch(pa0 + 32, 0, 3);
            __builtin_prefetch(pl0 + 32, 0, 3);
            __builtin_prefetch(Wt + (size_t)(Nbase + l15) * K_DIM + k0 + 32, 0, 3);
        }

        #pragma unroll
        for (int mt = 0; mt < 2; ++mt)
            #pragma unroll
            for (int nt = 0; nt < 4; ++nt) {
                acc[mt][nt] = __builtin_amdgcn_wmma_f32_16x16x32_f16(
                    false, ah[mt].v, false, bf[nt].v, (short)0, acc[mt][nt],
                    false, false);
                acc[mt][nt] = __builtin_amdgcn_wmma_f32_16x16x32_f16(
                    false, al[mt].v, false, bf[nt].v, (short)0, acc[mt][nt],
                    false, false);
            }
    }

    // C/D layout: lanes 0-15 -> N=l15, VGPR r -> M=r ; lanes 16-31 -> M=r+8
    #pragma unroll
    for (int nt = 0; nt < 4; ++nt) {
        const int n  = Nbase + nt * 16 + l15;
        const float bv = bias[n];
        #pragma unroll
        for (int mt = 0; mt < 2; ++mt) {
            #pragma unroll
            for (int r = 0; r < 8; ++r) {
                const int m = Mbase + mt * 16 + lhalf * 8 + r;
                out[(size_t)m * N_COLS + n] = acc[mt][nt][r] + bv;
            }
        }
    }
}

// ---------------------------------------------------------------------------
extern "C" void kernel_launch(void* const* d_in, const int* in_sizes, int n_in,
                              void* d_out, int out_size, void* d_ws, size_t ws_size,
                              hipStream_t stream) {
    const float* a = (const float*)d_in[0];   // [8192, 2048] fp32
    const float* w = (const float*)d_in[1];   // [2048, 2048] fp32
    const float* b = (const float*)d_in[2];   // [2048] fp32
    float* out = (float*)d_out;               // [8192, 2048] fp32

    char* ws = (char*)d_ws;
    _Float16* wt  = (_Float16*)ws;                                   // 8 MB
    _Float16* ahi = (_Float16*)(ws + (size_t)K_DIM * N_COLS * 2);    // 32 MB
    _Float16* alo = ahi + (size_t)N_ROWS * K_DIM;                    // 32 MB

    dim3 tb(32, 8);
    dim3 tg(N_COLS / 32, K_DIM / 32);
    binarize_transpose_w<<<tg, tb, 0, stream>>>(w, wt);

    const int total = N_ROWS * K_DIM;
    split_a<<<total / 256, 256, 0, stream>>>(a, ahi, alo, total);

    dim3 gb(256);
    dim3 gg(N_COLS / 256, N_ROWS / 64);   // (8, 128)
    gemm_wmma<<<gg, gb, 0, stream>>>(ahi, alo, wt, b, out);
}
